// CrossSRA_42202348650882
// MI455X (gfx1250) — compile-verified, matrix operand order checked by reading, not run on hardware
//
#include <hip/hip_runtime.h>
#include <hip/hip_bf16.h>
#include <stdint.h>

// ---------------------------------------------------------------------------
// CDNA5 (gfx1250) wave32 WMMA types
// ---------------------------------------------------------------------------
typedef __attribute__((ext_vector_type(16))) __bf16 v16bf;
typedef __attribute__((ext_vector_type(8)))  float  v8f;
typedef int v4i_vs __attribute__((vector_size(16)));   // matches builtin param

union FragBF { v16bf v; uint4 u[2]; };

// Async global->LDS copy path (ASYNCcnt-tracked) if this toolchain exposes it.
#if defined(__has_builtin)
#if __has_builtin(__builtin_amdgcn_global_load_async_to_lds_b128)
#define ATHENA_ASYNC_LDS 1
#endif
#endif

__device__ __forceinline__ void panel_copy16(unsigned short* ldst,
                                             const unsigned short* gsrc) {
#ifdef ATHENA_ASYNC_LDS
  __builtin_amdgcn_global_load_async_to_lds_b128(
      (__attribute__((address_space(1))) v4i_vs*)gsrc,
      (__attribute__((address_space(3))) v4i_vs*)ldst, 0, 0);
#else
  *(uint4*)ldst = *(const uint4*)gsrc;
#endif
}

__device__ __forceinline__ void panel_fence() {
#ifdef ATHENA_ASYNC_LDS
#if __has_builtin(__builtin_amdgcn_s_wait_asynccnt)
  __builtin_amdgcn_s_wait_asynccnt(0);
#else
  asm volatile("s_wait_asynccnt 0x0" ::: "memory");
#endif
#endif
}

__device__ __forceinline__ unsigned short f2bf(float x) {
  unsigned u = __builtin_bit_cast(unsigned, x);
  unsigned r = (u + 0x7FFFu + ((u >> 16) & 1u)) >> 16;   // RNE
  return (unsigned short)r;
}

__device__ __forceinline__ v8f zero8() {
  v8f z = {0.f, 0.f, 0.f, 0.f, 0.f, 0.f, 0.f, 0.f};
  return z;
}

// 16-bit A/B fragment from a row-major row.  ISA layout (16-bit A 16x32):
// lanes 0-15 hold K = k0+[0..7] and k0+[16..23]; lanes 16-31 hold K =
// k0+[8..15] and k0+[24..31].  The B-fragment of W^T is the identical
// pattern applied to rows of W.
__device__ __forceinline__ v16bf load_frag_row(const unsigned short* rowp,
                                               int k0, int hi) {
  FragBF f;
  const unsigned short* p = rowp + k0 + hi * 8;
  f.u[0] = *reinterpret_cast<const uint4*>(p);
  f.u[1] = *reinterpret_cast<const uint4*>(p + 16);
  return f.v;
}

__device__ __forceinline__ v8f wmma_bf16(v16bf a, v16bf b, v8f c) {
  return __builtin_amdgcn_wmma_f32_16x16x32_bf16(false, a, false, b,
                                                 (short)0, c, false, false);
}

// ---------------------------------------------------------------------------
// Elementwise helpers
// ---------------------------------------------------------------------------
__global__ void cvt_bf16_kernel(const float* __restrict__ in,
                                unsigned short* __restrict__ out, int n4) {
  int i = blockIdx.x * 256 + threadIdx.x;
  if (i < n4) {
    float4 f = reinterpret_cast<const float4*>(in)[i];
    ushort4 o;
    o.x = f2bf(f.x); o.y = f2bf(f.y); o.z = f2bf(f.z); o.w = f2bf(f.w);
    reinterpret_cast<ushort4*>(out)[i] = o;
  }
}

// srW [O=512][C=512][8][8] -> Wp [O][p=kh*8+kw][C]  (bf16): SR conv becomes a
// GEMM with K = 64*512 and contiguous fragments on both sides.
__global__ void pack_srw_kernel(const float* __restrict__ srW,
                                unsigned short* __restrict__ Wp) {
  int i = blockIdx.x * 256 + threadIdx.x;     // (o*64 + p)*512 + c
  int c = i & 511;
  int p = (i >> 9) & 63;
  int o = i >> 15;
  Wp[i] = f2bf(srW[((size_t)o * 512 + c) * 64 + p]);
}

// v [B*64, 512] bf16 -> vT [B,8,64(d),64(m)] bf16 so the P@V GEMM has
// row-major-contiguous B-fragments.
__global__ void transpose_v_kernel(const unsigned short* __restrict__ v,
                                   unsigned short* __restrict__ vT) {
  int i = blockIdx.x * 256 + threadIdx.x;     // ((b*8+h)*64+d)*64+m
  int m  = i & 63;
  int dd = (i >> 6) & 63;
  int h  = (i >> 12) & 7;
  int b  = i >> 15;
  vT[i] = v[((size_t)b * 64 + m) * 512 + h * 64 + dd];
}

// tokens_raw f32 [512,512]: add conv bias, LayerNorm, emit bf16 tokens.
__global__ void bias_ln_kernel(const float* __restrict__ raw,
                               const float* __restrict__ srb,
                               const float* __restrict__ g,
                               const float* __restrict__ bt,
                               unsigned short* __restrict__ tok) {
  int row = blockIdx.x;
  int t = threadIdx.x;                        // 256 threads, C = 512
  const float* rp = raw + (size_t)row * 512;
  float v0 = rp[t] + srb[t];
  float v1 = rp[t + 256] + srb[t + 256];
  __shared__ float red[8];
  float s = v0 + v1;
  #pragma unroll
  for (int o = 16; o > 0; o >>= 1) s += __shfl_xor(s, o, 32);
  if ((t & 31) == 0) red[t >> 5] = s;
  __syncthreads();
  float tot = 0.f;
  #pragma unroll
  for (int i = 0; i < 8; ++i) tot += red[i];
  float mean = tot * (1.f / 512.f);
  __syncthreads();
  float d0 = v0 - mean, d1 = v1 - mean;
  float q = d0 * d0 + d1 * d1;
  #pragma unroll
  for (int o = 16; o > 0; o >>= 1) q += __shfl_xor(q, o, 32);
  if ((t & 31) == 0) red[t >> 5] = q;
  __syncthreads();
  float vt = 0.f;
  #pragma unroll
  for (int i = 0; i < 8; ++i) vt += red[i];
  float rstd = rsqrtf(vt * (1.f / 512.f) + 1e-5f);
  unsigned short* op = tok + (size_t)row * 512;
  op[t]       = f2bf(d0 * rstd * g[t] + bt[t]);
  op[t + 256] = f2bf(d1 * rstd * g[t + 256] + bt[t + 256]);
}

// ---------------------------------------------------------------------------
// Pipelined WMMA GEMM:  Y[R,512] = X[R,K](bf16) @ W[512,K](bf16)^T (+ bias)
//
// Block = 256 threads (8 waves), 128x128 output tile.  The 128x32 W panel is
// staged in LDS (double-buffered, async global->LDS copy); each wave owns a
// 16x128 strip.  Per k-step: issue next panel + next A-frag, preload all 8
// B-fragments from LDS, then fire 8 back-to-back WMMAs.
//
// SR=1: A rows gather the (kh,kw) patch rows of x_ir (K = 64*512); within a
// 512-channel p-block the fragment loads stay contiguous 16B vectors.
// ---------------------------------------------------------------------------
template <int SR, int OUT_BF16>
__global__ __launch_bounds__(256) void gemm_wmma_kernel(
    const unsigned short* __restrict__ X,
    const unsigned short* __restrict__ W,
    const float* __restrict__ bias,
    void* __restrict__ Yv, int K) {
  const int NC = 512;
  int t = threadIdx.x;
  int lane = t & 31, wave = t >> 5;
  int hi = lane >> 4, ln = lane & 15;
  int row0 = blockIdx.y * 128 + wave * 16;
  int col0 = blockIdx.x * 128;

  __shared__ __align__(16) unsigned short wpan[2][128 * 32];

  // A-row addressing (SR path folds the patch gather into the row pointer).
  int arow = row0 + ln;
  int bidx = arow >> 6, m = arow & 63;
  int py = m >> 3, px = m & 7;
  size_t abase = SR ? ((size_t)bidx * 4096 + (size_t)py * 512 + (size_t)px * 8) * 512
                    : (size_t)arow * (size_t)K;

  auto a_ptr = [&](int k) -> const unsigned short* {
    if (SR) {
      int p = k >> 9, c0 = k & 511;
      return X + abase + ((size_t)(p >> 3) * 64 + (p & 7)) * 512 + c0;
    }
    return X + abase + k;
  };

  // Cooperative 128x32 panel stage: 512 uint4, 2 per thread.
  auto stage_panel = [&](int buf, int k) {
    for (int i = t; i < 512; i += 256) {
      int row = i >> 2, seg = i & 3;
      panel_copy16(&wpan[buf][row * 32 + seg * 8],
                   W + (size_t)(col0 + row) * K + k + seg * 8);
    }
  };

  const int nsteps = K / 32;

  stage_panel(0, 0);
  v16bf a_cur = load_frag_row(a_ptr(0), 0, hi);
  panel_fence();
  __syncthreads();

  v8f acc[8];
  #pragma unroll
  for (int j = 0; j < 8; ++j) acc[j] = zero8();

  int buf = 0;
  for (int step = 0; step < nsteps; ++step) {
    // Issue next panel + next A-fragment before touching this step's data.
    v16bf a_nxt = a_cur;
    if (step + 1 < nsteps) {
      int kn = (step + 1) * 32;
      stage_panel(buf ^ 1, kn);
      a_nxt = load_frag_row(a_ptr(kn), 0, hi);
    }

    // Preload all 8 B-fragments (ds_load clause), then 8 WMMAs back-to-back.
    v16bf bfrag[8];
    #pragma unroll
    for (int j = 0; j < 8; ++j) {
      FragBF f;
      const unsigned short* p = &wpan[buf][(j * 16 + ln) * 32 + hi * 8];
      f.u[0] = *reinterpret_cast<const uint4*>(p);
      f.u[1] = *reinterpret_cast<const uint4*>(p + 16);
      bfrag[j] = f.v;
    }
    #pragma unroll
    for (int j = 0; j < 8; ++j)
      acc[j] = wmma_bf16(a_cur, bfrag[j], acc[j]);

    panel_fence();
    __syncthreads();
    a_cur = a_nxt;
    buf ^= 1;
  }

  #pragma unroll
  for (int j = 0; j < 8; ++j) {
    int col = col0 + j * 16 + ln;
    float bv = bias ? bias[col] : 0.f;
    #pragma unroll
    for (int r = 0; r < 8; ++r) {   // D layout: lane row = r + hi*8
      size_t idx = (size_t)(row0 + hi * 8 + r) * NC + col;
      float val = acc[j][r] + bv;
      if (OUT_BF16) ((unsigned short*)Yv)[idx] = f2bf(val);
      else          ((float*)Yv)[idx] = val;
    }
  }
}

// ---------------------------------------------------------------------------
// Fused SRA attention per (b,h): S = Q K^T * scale, softmax over M=64,
// O = P V.  K and V^T staged in LDS; P converted D-layout -> A-layout through
// LDS.  Grid (32, 8, 8) = (N/128, heads, batch); 256 threads = 8 waves.
// ---------------------------------------------------------------------------
__global__ __launch_bounds__(256) void attn_kernel(
    const unsigned short* __restrict__ Q,
    const unsigned short* __restrict__ Kt,
    const unsigned short* __restrict__ Vt,
    unsigned short* __restrict__ Ctx) {
  int b = blockIdx.z, h = blockIdx.y, blk = blockIdx.x;
  int lane = threadIdx.x & 31, wave = threadIdx.x >> 5;
  int hi = lane >> 4, ln = lane & 15;

  __shared__ __align__(16) unsigned short k_s[64 * 64];    // [m][d]
  __shared__ __align__(16) unsigned short vT_s[64 * 64];   // [d][m]
  __shared__ __align__(16) unsigned short p_s[8][16 * 64]; // per-wave P tile

  for (int i = threadIdx.x; i < 512; i += 256) {
    int m = i >> 3;
    int c8 = (i & 7) * 8;
    *(uint4*)&k_s[m * 64 + c8] =
        *(const uint4*)&Kt[((size_t)b * 64 + m) * 512 + h * 64 + c8];
    *(uint4*)&vT_s[m * 64 + c8] =
        *(const uint4*)&Vt[(((size_t)b * 8 + h) * 64 + m) * 64 + c8];
  }
  __syncthreads();

  int n0 = blk * 128 + wave * 16;

  v8f s[4];
  #pragma unroll
  for (int j = 0; j < 4; ++j) s[j] = zero8();
  const unsigned short* qrow = Q + ((size_t)b * 4096 + n0 + ln) * 512 + h * 64;
  #pragma unroll
  for (int kk = 0; kk < 64; kk += 32) {
    v16bf a = load_frag_row(qrow, kk, hi);
    #pragma unroll
    for (int j = 0; j < 4; ++j) {
      v16bf bb = load_frag_row(&k_s[(j * 16 + ln) * 64], kk, hi);
      s[j] = wmma_bf16(a, bb, s[j]);
    }
  }

  // Softmax over 64 cols.  D layout: lane holds col = j*16+ln, rows hi*8+r;
  // each row's 64 values live in one 16-lane half -> butterfly over 1,2,4,8.
  const float scale = 0.125f;   // d^-0.5, d = 64
  #pragma unroll
  for (int r = 0; r < 8; ++r) {
    float sv[4];
    float mx = -1e30f;
    #pragma unroll
    for (int j = 0; j < 4; ++j) { sv[j] = s[j][r] * scale; mx = fmaxf(mx, sv[j]); }
    #pragma unroll
    for (int o = 1; o < 16; o <<= 1) mx = fmaxf(mx, __shfl_xor(mx, o, 32));
    float sum = 0.f;
    #pragma unroll
    for (int j = 0; j < 4; ++j) { sv[j] = __expf(sv[j] - mx); sum += sv[j]; }
    #pragma unroll
    for (int o = 1; o < 16; o <<= 1) sum += __shfl_xor(sum, o, 32);
    float inv = 1.0f / sum;
    #pragma unroll
    for (int j = 0; j < 4; ++j)
      p_s[wave][(hi * 8 + r) * 64 + j * 16 + ln] = f2bf(sv[j] * inv);
  }

  v8f o[4];
  #pragma unroll
  for (int j = 0; j < 4; ++j) o[j] = zero8();
  #pragma unroll
  for (int kk = 0; kk < 64; kk += 32) {
    v16bf a = load_frag_row(&p_s[wave][ln * 64], kk, hi);
    #pragma unroll
    for (int j = 0; j < 4; ++j) {
      v16bf bb = load_frag_row(&vT_s[(j * 16 + ln) * 64], kk, hi);
      o[j] = wmma_bf16(a, bb, o[j]);
    }
  }

  #pragma unroll
  for (int j = 0; j < 4; ++j)
    #pragma unroll
    for (int r = 0; r < 8; ++r)
      Ctx[((size_t)b * 4096 + n0 + hi * 8 + r) * 512 + h * 64 + j * 16 + ln] =
          f2bf(o[j][r]);
}

// ---------------------------------------------------------------------------
// Host orchestration
// ---------------------------------------------------------------------------
extern "C" void kernel_launch(void* const* d_in, const int* in_sizes, int n_in,
                              void* d_out, int out_size, void* d_ws, size_t ws_size,
                              hipStream_t stream) {
  (void)in_sizes; (void)n_in; (void)out_size; (void)ws_size;
  const float* x_vis = (const float*)d_in[0];
  const float* x_ir  = (const float*)d_in[1];
  const float* qW    = (const float*)d_in[2];
  const float* qb    = (const float*)d_in[3];
  const float* kW    = (const float*)d_in[4];
  const float* kb    = (const float*)d_in[5];
  const float* vW    = (const float*)d_in[6];
  const float* vb    = (const float*)d_in[7];
  const float* projW = (const float*)d_in[8];
  const float* projb = (const float*)d_in[9];
  const float* srW   = (const float*)d_in[10];
  const float* srb   = (const float*)d_in[11];
  const float* ln_g  = (const float*)d_in[12];
  const float* ln_b  = (const float*)d_in[13];

  const size_t BNC = (size_t)8 * 4096 * 512;   // 16,777,216
  const size_t CC  = (size_t)512 * 512;

  char* w = (char*)d_ws;
  size_t off = 0;
  auto take = [&](size_t bytes) {
    char* p = w + off;
    off += (bytes + 255) & ~(size_t)255;
    return p;
  };
  unsigned short* xvis_bf  = (unsigned short*)take(BNC * 2);
  unsigned short* xir_bf   = (unsigned short*)take(BNC * 2);
  unsigned short* qW_bf    = (unsigned short*)take(CC * 2);
  unsigned short* kW_bf    = (unsigned short*)take(CC * 2);
  unsigned short* vW_bf    = (unsigned short*)take(CC * 2);
  unsigned short* projW_bf = (unsigned short*)take(CC * 2);
  unsigned short* srWp_bf  = (unsigned short*)take(CC * 64 * 2); // [O][64][C]
  unsigned short* q_bf     = (unsigned short*)take(BNC * 2);
  float*          tok_raw  = (float*)take(CC * 4);               // [512,512]
  unsigned short* tok_bf   = (unsigned short*)take(CC * 2);
  unsigned short* k_bf     = (unsigned short*)take(CC * 2);
  unsigned short* v_bf     = (unsigned short*)take(CC * 2);
  unsigned short* vT_bf    = (unsigned short*)take(CC * 2);
  unsigned short* ctx_bf   = (unsigned short*)take(BNC * 2);

  // 1. bf16 conversions + weight repack
  cvt_bf16_kernel<<<(int)(BNC / 4 / 256), 256, 0, stream>>>(x_vis, xvis_bf, (int)(BNC / 4));
  cvt_bf16_kernel<<<(int)(BNC / 4 / 256), 256, 0, stream>>>(x_ir,  xir_bf,  (int)(BNC / 4));
  cvt_bf16_kernel<<<(int)(CC / 4 / 256), 256, 0, stream>>>(qW,    qW_bf,    (int)(CC / 4));
  cvt_bf16_kernel<<<(int)(CC / 4 / 256), 256, 0, stream>>>(kW,    kW_bf,    (int)(CC / 4));
  cvt_bf16_kernel<<<(int)(CC / 4 / 256), 256, 0, stream>>>(vW,    vW_bf,    (int)(CC / 4));
  cvt_bf16_kernel<<<(int)(CC / 4 / 256), 256, 0, stream>>>(projW, projW_bf, (int)(CC / 4));
  pack_srw_kernel<<<(int)(CC * 64 / 256), 256, 0, stream>>>(srW, srWp_bf);

  // 2. Q projection (bf16 out)
  gemm_wmma_kernel<0, 1><<<dim3(4, 256), 256, 0, stream>>>(xvis_bf, qW_bf, qb, q_bf, 512);

  // 3. SR conv as GEMM -> raw tokens f32 [512,512]
  gemm_wmma_kernel<1, 0><<<dim3(4, 4), 256, 0, stream>>>(xir_bf, srWp_bf, nullptr, tok_raw, 32768);

  // 4. +srb, LayerNorm -> bf16 tokens
  bias_ln_kernel<<<512, 256, 0, stream>>>(tok_raw, srb, ln_g, ln_b, tok_bf);

  // 5. K, V projections (bf16 out)
  gemm_wmma_kernel<0, 1><<<dim3(4, 4), 256, 0, stream>>>(tok_bf, kW_bf, kb, k_bf, 512);
  gemm_wmma_kernel<0, 1><<<dim3(4, 4), 256, 0, stream>>>(tok_bf, vW_bf, vb, v_bf, 512);

  // 6. V -> V^T [B,h,d,M]
  transpose_v_kernel<<<1024, 256, 0, stream>>>(v_bf, vT_bf);

  // 7. Fused attention -> ctx bf16 [B,N,C]
  attn_kernel<<<dim3(32, 8, 8), 256, 0, stream>>>(q_bf, k_bf, vT_bf, ctx_bf);

  // 8. Output projection -> d_out f32 [B,N,C]
  gemm_wmma_kernel<0, 0><<<dim3(4, 256), 256, 0, stream>>>(ctx_bf, projW_bf, projb, d_out, 512);
}
